// LSTMBaseline_72593537237489
// MI455X (gfx1250) — matrix operand order repaired and checked by reading, not compiled, not run on hardware
//
#include <hip/hip_runtime.h>
#include <hip/hip_bf16.h>

typedef __bf16 v16bf __attribute__((ext_vector_type(16)));
typedef __bf16 v8bf  __attribute__((ext_vector_type(8)));
typedef float  v8f   __attribute__((ext_vector_type(8)));

#define Bsz   2048
#define Tsz   512
#define Hsz   128
#define FFsz  64
#define HS    136      // padded LDS row stride (elements): 272B -> conflict-free A-fragment loads
#define TILE_M 16
#define NTH    128     // 4 waves -> 1 wave per SIMD -> full ~1024 VGPR budget, no spills

__device__ __forceinline__ float sigmoid_f(float x) {
    return __builtin_amdgcn_rcpf(1.0f + __expf(-x));
}
__device__ __forceinline__ float tanh_f(float x) {
    float e = __expf(2.0f * x);
    return (e - 1.0f) * __builtin_amdgcn_rcpf(e + 1.0f);
}

// A-fragment (16x32 bf16) from an LDS h buffer per ISA 7.12.2:
// lane = 16*kh + M ; elements e<8 -> K = kc*32 + kh*8 + e ; e>=8 -> K = kc*32 + 16 + kh*8 + (e-8)
__device__ __forceinline__ v16bf load_afrag(const __bf16* h, int nlo, int kh, int kc) {
    const __bf16* p = h + nlo * HS + kc * 32 + kh * 8;   // 16B aligned
    v8bf lo = *(const v8bf*)(p);
    v8bf hi = *(const v8bf*)(p + 16);
    return __builtin_shufflevector(lo, hi, 0,1,2,3,4,5,6,7,8,9,10,11,12,13,14,15);
}

#define WMMA_BF16(A, B, C) \
    __builtin_amdgcn_wmma_f32_16x16x32_bf16(false, (A), false, (B), (short)0, (C), false, false)

__global__ __launch_bounds__(NTH, 1) void lstm2_fused_kernel(
    const float* __restrict__ hr,   const float* __restrict__ gl,
    const float* __restrict__ w_ih0,const float* __restrict__ w_hh0,
    const float* __restrict__ b_ih0,const float* __restrict__ b_hh0,
    const float* __restrict__ w_ih1,const float* __restrict__ w_hh1,
    const float* __restrict__ b_ih1,const float* __restrict__ b_hh1,
    const float* __restrict__ w1,   const float* __restrict__ b1,
    const float* __restrict__ w2,   const float* __restrict__ b2,
    float* __restrict__ out)
{
    __shared__ float  s_hr[TILE_M * Tsz];          // 32 KB
    __shared__ float  s_gl[TILE_M * Tsz];          // 32 KB
    __shared__ __bf16 s_h0[2][TILE_M * HS];        // ping-pong layer-0 hidden (bf16)
    __shared__ __bf16 s_h1[2][TILE_M * HS];        // ping-pong layer-1 hidden (bf16)
    __shared__ float  s_hid[TILE_M * FFsz];        // head scratch

    const int tid  = threadIdx.x;
    const int lane = tid & 31;
    const int wave = tid >> 5;    // 0..3
    const int nlo  = lane & 15;   // N within tile / A-row M
    const int kh   = lane >> 4;   // lane half
    const int u0   = wave * 32;   // hidden-unit base owned by this wave (32 units)
    const int bb   = blockIdx.x * TILE_M;

    // ---- stage input time series for this batch tile (coalesced over t) ----
    for (int i = tid; i < TILE_M * Tsz; i += NTH) {
        int m = i >> 9, t = i & (Tsz - 1);
        s_hr[i] = hr[(bb + m) * Tsz + t];
        s_gl[i] = gl[(bb + m) * Tsz + t];
    }
    for (int i = tid; i < TILE_M * HS; i += NTH) {
        s_h0[0][i] = (__bf16)0.0f; s_h0[1][i] = (__bf16)0.0f;
        s_h1[0][i] = (__bf16)0.0f; s_h1[1][i] = (__bf16)0.0f;
    }

    // ---- load + convert weight B-fragments into registers (once) ----
    // This wave owns N-tiles j = g*2 + s : gate g (i,f,g,o), unit subtile s.
    // B 32x16 bf16 layout: lane N = nlo, element e -> K = kc*32 + kh*8 + (e<8 ? e : e+8) ; value = W[n_global, K]
    v16bf W0[8][4], W1[8][4], W2[8][4];   // [ntile][k-chunk] : 768 VGPRs total
    #pragma unroll
    for (int j = 0; j < 8; ++j) {
        const int g = j >> 1, s = j & 1;
        const int row = g * Hsz + u0 + s * 16 + nlo;   // gate column in [4H, H] weights
        const float* p0 = w_hh0 + row * Hsz;
        const float* p1 = w_ih1 + row * Hsz;
        const float* p2 = w_hh1 + row * Hsz;
        #pragma unroll
        for (int kc = 0; kc < 4; ++kc) {
            const int k0 = kc * 32 + kh * 8;
            v16bf a, b, c;
            #pragma unroll
            for (int e = 0; e < 16; ++e) {
                a[e] = (__bf16)p0[k0 + (e < 8 ? e : e + 8)];
                b[e] = (__bf16)p1[k0 + (e < 8 ? e : e + 8)];
                c[e] = (__bf16)p2[k0 + (e < 8 ? e : e + 8)];
            }
            W0[j][kc] = a; W1[j][kc] = b; W2[j][kc] = c;
        }
    }

    // per-lane layer-0 input weights (input dim is only 2) and fused biases
    float wa[8], wb[8], bs0[8], bs1[8];
    #pragma unroll
    for (int j = 0; j < 8; ++j) {
        const int g = j >> 1, s = j & 1;
        const int row = g * Hsz + u0 + s * 16 + nlo;
        wa[j]  = w_ih0[row * 2 + 0];
        wb[j]  = w_ih0[row * 2 + 1];
        bs0[j] = b_ih0[row] + b_hh0[row];
        bs1[j] = b_ih1[row] + b_hh1[row];
    }

    // cell states in registers: lane -> (m = r + 8*kh, unit = u0 + 16*s + nlo)
    v8f c0[2] = {}; v8f c1[2] = {};
    __syncthreads();

    int cur = 0;
    for (int t = 0; t < Tsz; ++t) {
        const int nxt = cur ^ 1;

        // ================= layer 0 =================
        v8f acc[8];
        #pragma unroll
        for (int r = 0; r < 8; ++r) {
            const int m = r + 8 * kh;
            const float xh = s_hr[m * Tsz + t];
            const float xg = s_gl[m * Tsz + t];
            #pragma unroll
            for (int j = 0; j < 8; ++j)
                acc[j][r] = bs0[j] + wa[j] * xh + wb[j] * xg;
        }
        #pragma unroll
        for (int kc = 0; kc < 4; ++kc) {
            v16bf af = load_afrag(s_h0[cur], nlo, kh, kc);
            #pragma unroll
            for (int j = 0; j < 8; ++j)
                acc[j] = WMMA_BF16(af, W0[j][kc], acc[j]);
        }
        #pragma unroll
        for (int s = 0; s < 2; ++s) {
            #pragma unroll
            for (int r = 0; r < 8; ++r) {
                const float ig = sigmoid_f(acc[0 + s][r]);
                const float fg = sigmoid_f(acc[2 + s][r]);
                const float gg = tanh_f   (acc[4 + s][r]);
                const float og = sigmoid_f(acc[6 + s][r]);
                const float c  = fg * c0[s][r] + ig * gg;
                c0[s][r] = c;
                s_h0[nxt][(r + 8 * kh) * HS + u0 + s * 16 + nlo] = (__bf16)(og * tanh_f(c));
            }
        }
        __syncthreads();   // publish h0[t] to all waves

        // ================= layer 1 =================
        #pragma unroll
        for (int j = 0; j < 8; ++j)
            #pragma unroll
            for (int r = 0; r < 8; ++r) acc[j][r] = bs1[j];
        #pragma unroll
        for (int kc = 0; kc < 4; ++kc) {           // input projection: h0[t] @ w_ih1^T
            v16bf af = load_afrag(s_h0[nxt], nlo, kh, kc);
            #pragma unroll
            for (int j = 0; j < 8; ++j)
                acc[j] = WMMA_BF16(af, W1[j][kc], acc[j]);
        }
        #pragma unroll
        for (int kc = 0; kc < 4; ++kc) {           // recurrent: h1[t-1] @ w_hh1^T
            v16bf af = load_afrag(s_h1[cur], nlo, kh, kc);
            #pragma unroll
            for (int j = 0; j < 8; ++j)
                acc[j] = WMMA_BF16(af, W2[j][kc], acc[j]);
        }
        #pragma unroll
        for (int s = 0; s < 2; ++s) {
            #pragma unroll
            for (int r = 0; r < 8; ++r) {
                const float ig = sigmoid_f(acc[0 + s][r]);
                const float fg = sigmoid_f(acc[2 + s][r]);
                const float gg = tanh_f   (acc[4 + s][r]);
                const float og = sigmoid_f(acc[6 + s][r]);
                const float c  = fg * c1[s][r] + ig * gg;
                c1[s][r] = c;
                s_h1[nxt][(r + 8 * kh) * HS + u0 + s * 16 + nlo] = (__bf16)(og * tanh_f(c));
            }
        }
        __syncthreads();   // publish h1[t]
        cur = nxt;
    }

    // ================= regression head (negligible cost) =================
    for (int i = tid; i < TILE_M * FFsz; i += NTH) {
        const int m = i >> 6, f = i & (FFsz - 1);
        float s = b1[f];
        const __bf16* hp = &s_h1[cur][m * HS];
        #pragma unroll 8
        for (int k = 0; k < Hsz; ++k) s += (float)hp[k] * w1[f * Hsz + k];
        s_hid[i] = fmaxf(s, 0.0f);
    }
    __syncthreads();
    for (int i = tid; i < TILE_M * 2; i += NTH) {
        const int m = i >> 1, o = i & 1;
        float s = b2[o];
        #pragma unroll 8
        for (int f = 0; f < FFsz; ++f) s += s_hid[m * FFsz + f] * w2[o * FFsz + f];
        out[(bb + m) * 2 + o] = s;
    }
}

extern "C" void kernel_launch(void* const* d_in, const int* in_sizes, int n_in,
                              void* d_out, int out_size, void* d_ws, size_t ws_size,
                              hipStream_t stream) {
    (void)in_sizes; (void)n_in; (void)d_ws; (void)ws_size; (void)out_size;
    lstm2_fused_kernel<<<Bsz / TILE_M, NTH, 0, stream>>>(
        (const float*)d_in[0],  (const float*)d_in[1],
        (const float*)d_in[2],  (const float*)d_in[3],
        (const float*)d_in[4],  (const float*)d_in[5],
        (const float*)d_in[6],  (const float*)d_in[7],
        (const float*)d_in[8],  (const float*)d_in[9],
        (const float*)d_in[10], (const float*)d_in[11],
        (const float*)d_in[12], (const float*)d_in[13],
        (float*)d_out);
}
